// GraphConvLayer_63591285784877
// MI455X (gfx1250) — matrix operand order repaired and checked by reading, not compiled, not run on hardware
//
#include <hip/hip_runtime.h>

#define B_ 4
#define N_ 2048
#define F_ 128

#define BM 32      // output rows per block
#define KC 32      // K chunk (one bf16-WMMA K step)
#define AF_LD 36   // LDS stride (fp32 elems) for raw A tile: 144B rows, 16B-aligned runs
#define T_LD 40    // LDS stride (bf16 elems) for 32-wide bf16 tiles: 80B rows
#define A2_LD 136  // LDS stride for GEMM2 resident A tile (272B rows)

typedef __attribute__((ext_vector_type(8)))  float v8f;
typedef __attribute__((ext_vector_type(8)))  __bf16 v8bf;
typedef __attribute__((ext_vector_type(16))) __bf16 v16bf;
typedef __attribute__((ext_vector_type(4)))  unsigned short us4;
typedef __attribute__((ext_vector_type(16))) unsigned short us16;
typedef __attribute__((ext_vector_type(4)))  int v4i;

#if defined(__AMDGCN__) &&                                        \
    __has_builtin(__builtin_amdgcn_global_load_async_to_lds_b128) && \
    __has_builtin(__builtin_amdgcn_s_wait_asynccnt)
#define USE_ASYNC 1
#else
#define USE_ASYNC 0
#endif

__device__ __forceinline__ v8f wmma_bf16(v16bf a, v16bf b, v8f c) {
  // probe-confirmed: (neg_a, A, neg_b, B, c_mod, C, reuse_a, reuse_b)
  return __builtin_amdgcn_wmma_f32_16x16x32_bf16(false, a, false, b, (short)0,
                                                 c, false, false);
}

// fp32 -> bf16 hi/lo split (truncation; lo captures next 8 mantissa bits)
__device__ __forceinline__ unsigned short bf_hi(float v) {
  return (unsigned short)(__float_as_uint(v) >> 16);
}
__device__ __forceinline__ unsigned short bf_lo(float v) {
  const unsigned int u = __float_as_uint(v);
  const float rem = v - __uint_as_float(u & 0xffff0000u);
  return (unsigned short)(__float_as_uint(rem) >> 16);
}

__device__ __forceinline__ void split4(float4 v, us4& h, us4& l) {
  h[0] = bf_hi(v.x); l[0] = bf_lo(v.x);
  h[1] = bf_hi(v.y); l[1] = bf_lo(v.y);
  h[2] = bf_hi(v.z); l[2] = bf_lo(v.z);
  h[3] = bf_hi(v.w); l[3] = bf_lo(v.w);
}

union v16u {
  us16 u;
  v16bf b;
};

// 16 fp32 from LDS (two 32B runs) -> hi/lo bf16 fragments, split in VGPRs
__device__ __forceinline__ void split_frag16(const float* base, int off0,
                                             int off1, v16bf& ah, v16bf& al) {
  float f[16];
  *(float4*)(f + 0)  = *(const float4*)(base + off0);
  *(float4*)(f + 4)  = *(const float4*)(base + off0 + 4);
  *(float4*)(f + 8)  = *(const float4*)(base + off1);
  *(float4*)(f + 12) = *(const float4*)(base + off1 + 4);
  us16 hu, lu;
#pragma unroll
  for (int e = 0; e < 16; ++e) {
    hu[e] = bf_hi(f[e]);
    lu[e] = bf_lo(f[e]);
  }
  v16u th, tl;
  th.u = hu; ah = th.b;
  tl.u = lu; al = tl.b;
}

// Two contiguous 16B LDS reads -> one 16-element bf16 fragment
__device__ __forceinline__ v16bf load_frag(const unsigned short* p, int run0,
                                           int run1) {
  v8bf x0 = *(const v8bf*)(p + run0);
  v8bf x1 = *(const v8bf*)(p + run1);
  return __builtin_shufflevector(x0, x1, 0, 1, 2, 3, 4, 5, 6, 7, 8, 9, 10, 11,
                                 12, 13, 14, 15);
}

// ---------------------------------------------------------------------------
// Kernel 1: dis[b,n] = rsqrt(max(rowsum(adj[b,n,:]), 1e-6))
// ---------------------------------------------------------------------------
__global__ __launch_bounds__(256) void gcn_degree_kernel(
    const float* __restrict__ adj, float* __restrict__ dis) {
  __shared__ float red[256];
  const int row = blockIdx.x;  // 0 .. B_*N_-1
  const float* a = adj + (size_t)row * N_;
  float s = 0.f;
#pragma unroll
  for (int i = 0; i < N_ / 256; ++i) s += a[threadIdx.x + i * 256];
  red[threadIdx.x] = s;
  __syncthreads();
  for (int off = 128; off > 0; off >>= 1) {
    if ((int)threadIdx.x < off) red[threadIdx.x] += red[threadIdx.x + off];
    __syncthreads();
  }
  if (threadIdx.x == 0) dis[row] = rsqrtf(fmaxf(red[0], 1e-6f));
}

// ---------------------------------------------------------------------------
// Kernel 2: agg[b,n,f] = dis[b,n] * sum_m adj[b,n,m] * (dis[b,m] * x[b,m,f])
// 128 threads (4 waves), 32x128 output tile, bf16x3 split-precision WMMA.
// A tile staged raw fp32 via async global->LDS DMA; split at fragment load.
// ---------------------------------------------------------------------------
__global__ __launch_bounds__(128) void gcn_agg_kernel(
    const float* __restrict__ adj, const float* __restrict__ x,
    const float* __restrict__ dis, float* __restrict__ agg) {
  __shared__ __align__(16) float As[BM * AF_LD];              // raw adj tile
  __shared__ __align__(16) unsigned short Bt_hi[F_ * T_LD];   // Bt[n][k] hi
  __shared__ __align__(16) unsigned short Bt_lo[F_ * T_LD];   // Bt[n][k] lo

  const int batch = blockIdx.x >> 6;  // 64 row-tiles per batch
  const int row0 = (blockIdx.x & 63) * BM;
  const float* adjb = adj + (size_t)batch * N_ * N_;
  const float* xb = x + (size_t)batch * N_ * F_;
  const float* disb = dis + batch * N_;

  const int tid = threadIdx.x;
  const int lane = tid & 31;
  const int wv = tid >> 5;    // 0..3
  const int h = lane >> 4;    // lane half (WMMA K split)
  const int lo = lane & 15;
  const int tr = wv >> 1;     // wave's 16-row stripe (0..1)
  const int cg = wv & 1;      // wave's 64-col group (0..1)

  const v8f zero = {0.f, 0.f, 0.f, 0.f, 0.f, 0.f, 0.f, 0.f};
  v8f acc[4] = {zero, zero, zero, zero};

  for (int k0 = 0; k0 < N_; k0 += KC) {
    // Stage A tile raw: adj[row0..+31][k0..+31] (256 x 16B)
#pragma unroll
    for (int j = 0; j < 2; ++j) {
      const int q = tid + j * 128;
      const int r = q >> 3, c4 = q & 7;
      const float* gsrc = adjb + (size_t)(row0 + r) * N_ + k0 + c4 * 4;
      float* ldst = As + r * AF_LD + c4 * 4;
#if USE_ASYNC
      __builtin_amdgcn_global_load_async_to_lds_b128(
          (__attribute__((address_space(1))) v4i*)gsrc,
          (__attribute__((address_space(3))) v4i*)ldst, 0, 0);
#else
      *(float4*)ldst = *(const float4*)gsrc;
#endif
    }
    // Stage B tile K-transposed with column scaling folded in:
    // Bt[n][k] = dis[k0+k] * x[k0+k][n]; thread tid owns column n = tid.
#pragma unroll
    for (int k = 0; k < KC; ++k) {
      const float v = xb[(size_t)(k0 + k) * F_ + tid] * disb[k0 + k];
      Bt_hi[tid * T_LD + k] = bf_hi(v);
      Bt_lo[tid * T_LD + k] = bf_lo(v);
    }
    // Prefetch next adjacency chunk (global_prefetch_b8)
    if (k0 + KC < N_)
      __builtin_prefetch(adjb + (size_t)(row0 + (tid & 31)) * N_ + k0 + KC, 0, 0);
#if USE_ASYNC
    __builtin_amdgcn_s_wait_asynccnt(0);
#endif
    __syncthreads();

    // One K=32 WMMA step per chunk; 3 products for split precision
    v16bf ah, al;
    split_frag16(As + (tr * 16 + lo) * AF_LD, 8 * h, 16 + 8 * h, ah, al);
#pragma unroll
    for (int ct = 0; ct < 4; ++ct) {
      const int n = cg * 64 + ct * 16 + lo;
      const v16bf bh = load_frag(Bt_hi + n * T_LD, 8 * h, 16 + 8 * h);
      const v16bf bl = load_frag(Bt_lo + n * T_LD, 8 * h, 16 + 8 * h);
      acc[ct] = wmma_bf16(ah, bh, acc[ct]);
      acc[ct] = wmma_bf16(al, bh, acc[ct]);
      acc[ct] = wmma_bf16(ah, bl, acc[ct]);
    }
    __syncthreads();
  }

  // Epilogue: row scaling by dis[n]; C/D layout: M = v + 8*h, N = lo
  float* aggb = agg + (size_t)batch * N_ * F_;
#pragma unroll
  for (int v = 0; v < 8; ++v) {
    const int r = tr * 16 + h * 8 + v;
    const float dr = disb[row0 + r];
#pragma unroll
    for (int ct = 0; ct < 4; ++ct) {
      const int n = cg * 64 + ct * 16 + lo;
      aggb[(size_t)(row0 + r) * F_ + n] = dr * acc[ct][v];
    }
  }
}

// ---------------------------------------------------------------------------
// Kernel 3: out[r,o] = sum_f agg[r,f] * W[o,f] + bias[o]   (in-place on d_out)
// B-matrix is W^T; W row-major IS the K-transposed LDS layout Bt[n][k].
// ---------------------------------------------------------------------------
__global__ __launch_bounds__(128) void gcn_out_kernel(
    const float* agg, const float* __restrict__ W,
    const float* __restrict__ bias, float* out) {
  __shared__ __align__(16) unsigned short As_hi[BM * A2_LD];  // agg tile hi
  __shared__ __align__(16) unsigned short As_lo[BM * A2_LD];  // agg tile lo
  __shared__ __align__(16) unsigned short Wc_hi[F_ * T_LD];   // W[:, f0..+31] hi
  __shared__ __align__(16) unsigned short Wc_lo[F_ * T_LD];   // W[:, f0..+31] lo

  const int row0 = blockIdx.x * BM;  // flattened over B_*N_ rows
  const int tid = threadIdx.x;
  const int lane = tid & 31;
  const int wv = tid >> 5;
  const int h = lane >> 4;
  const int lo = lane & 15;
  const int tr = wv >> 1;
  const int cg = wv & 1;

  // Stage full A tile once (K = 128), split to bf16 hi/lo
#pragma unroll
  for (int j = 0; j < 8; ++j) {
    const int q = tid + j * 128;  // 1024 float4 total
    const int r = q >> 5, c4 = q & 31;
    float4 v = *(const float4*)(agg + (size_t)(row0 + r) * F_ + c4 * 4);
    us4 hh, ll;
    split4(v, hh, ll);
    *(us4*)(As_hi + r * A2_LD + c4 * 4) = hh;
    *(us4*)(As_lo + r * A2_LD + c4 * 4) = ll;
  }

  const v8f zero = {0.f, 0.f, 0.f, 0.f, 0.f, 0.f, 0.f, 0.f};
  v8f acc[4] = {zero, zero, zero, zero};
  const unsigned short* Arh = As_hi + (tr * 16 + lo) * A2_LD;
  const unsigned short* Arl = As_lo + (tr * 16 + lo) * A2_LD;

  for (int f0 = 0; f0 < F_; f0 += KC) {
    // Stage W column-chunk: Wc[o][k] = W[o][f0+k] (already "transposed" form)
#pragma unroll
    for (int j = 0; j < 8; ++j) {
      const int q = tid + j * 128;  // 1024 float4 total
      const int o = q >> 3, c4 = q & 7;
      float4 v = *(const float4*)(W + (size_t)o * F_ + f0 + c4 * 4);
      us4 hh, ll;
      split4(v, hh, ll);
      *(us4*)(Wc_hi + o * T_LD + c4 * 4) = hh;
      *(us4*)(Wc_lo + o * T_LD + c4 * 4) = ll;
    }
    __syncthreads();

    const v16bf ah = load_frag(Arh, f0 + 8 * h, f0 + 16 + 8 * h);
    const v16bf al = load_frag(Arl, f0 + 8 * h, f0 + 16 + 8 * h);
#pragma unroll
    for (int ct = 0; ct < 4; ++ct) {
      const int n = cg * 64 + ct * 16 + lo;
      const v16bf bh = load_frag(Wc_hi + n * T_LD, 8 * h, 16 + 8 * h);
      const v16bf bl = load_frag(Wc_lo + n * T_LD, 8 * h, 16 + 8 * h);
      acc[ct] = wmma_bf16(ah, bh, acc[ct]);
      acc[ct] = wmma_bf16(al, bh, acc[ct]);
      acc[ct] = wmma_bf16(ah, bl, acc[ct]);
    }
    __syncthreads();
  }

#pragma unroll
  for (int v = 0; v < 8; ++v) {
    const int r = tr * 16 + h * 8 + v;
#pragma unroll
    for (int ct = 0; ct < 4; ++ct) {
      const int n = cg * 64 + ct * 16 + lo;
      out[(size_t)(row0 + r) * F_ + n] = acc[ct][v] + bias[n];
    }
  }
}

// ---------------------------------------------------------------------------
extern "C" void kernel_launch(void* const* d_in, const int* in_sizes, int n_in,
                              void* d_out, int out_size, void* d_ws,
                              size_t ws_size, hipStream_t stream) {
  (void)in_sizes; (void)n_in; (void)out_size; (void)ws_size;
  const float* x = (const float*)d_in[0];     // [B,N,F]
  const float* adj = (const float*)d_in[1];   // [B,N,N]
  const float* W = (const float*)d_in[2];     // [F_OUT,F_IN]
  const float* bias = (const float*)d_in[3];  // [F_OUT]
  float* out = (float*)d_out;                 // [B,N,F_OUT]
  float* dis = (float*)d_ws;                  // B_*N_ floats scratch

  gcn_degree_kernel<<<B_ * N_, 256, 0, stream>>>(adj, dis);
  // agg written into d_out, then the final GEMM rewrites d_out in place
  // (each block consumes its rows into LDS before overwriting them).
  gcn_agg_kernel<<<B_ * (N_ / BM), 128, 0, stream>>>(adj, x, dis, out);
  gcn_out_kernel<<<(B_ * N_) / BM, 128, 0, stream>>>(out, W, bias, out);
}
